// GlobalPointer_3831110828777
// MI455X (gfx1250) — matrix operand order repaired and checked by reading, not compiled, not run on hardware
//
#include <hip/hip_runtime.h>

typedef __attribute__((ext_vector_type(4)))  float   f4;
typedef __attribute__((ext_vector_type(8)))  float   v8f;
typedef __attribute__((ext_vector_type(4)))  __bf16  bf4;
typedef __attribute__((ext_vector_type(8)))  __bf16  bf8;
typedef __attribute__((ext_vector_type(16))) __bf16  v16bf;

// Problem constants
constexpr int HEADS = 12;
constexpr int HEAD  = 64;
constexpr int HID   = 768;
constexpr int NQK   = 1536;           // 2 * HEADS * HEAD
constexpr int BSZ   = 4;
constexpr int SEQ   = 1024;
constexpr size_t LOGITS_ELEMS = (size_t)BSZ * SEQ * SEQ * HEADS;   // 50,331,648
constexpr size_t QK_ELEMS     = (size_t)BSZ * SEQ * HEADS * HEAD;  //  3,145,728

// ln(10000)/32 for the RoPE inverse frequencies
#define ROPE_LN 0.28782313662425573f

__device__ __forceinline__ v16bf cat8(bf8 a, bf8 b) {
  return __builtin_shufflevector(a, b, 0,1,2,3,4,5,6,7,8,9,10,11,12,13,14,15);
}
__device__ __forceinline__ v8f wmma_bf16(v16bf a, v16bf b, v8f c) {
  return __builtin_amdgcn_wmma_f32_16x16x32_bf16(false, a, false, b, (short)0, c, false, false);
}

// ---------------------------------------------------------------------------
// Kernel 1: X = inputs @ W + b  (M=4096, N=1536, K=768), fp32 via split-bf16
// WMMA (hi/lo compensation, 3 products).  Epilogue: write org_q/org_k to
// d_out, RoPE-rotate in-register, spill rotated q,k to workspace (fp32).
// Block = 256 thr (8 waves); block tile 128(M) x 64(N); wave tile 16 x 64.
// ---------------------------------------------------------------------------
constexpr int ROWA = 40;   // padded LDS row (in bf16 elems): 80B, 16B-aligned,
constexpr int ROWB = 40;   // 20-bank stride -> 16 distinct banks across lanes

__global__ __launch_bounds__(256)
void qkv_rope_kernel(const float* __restrict__ Xin,
                     const float* __restrict__ Wm,
                     const float* __restrict__ bias,
                     float* __restrict__ org_q, float* __restrict__ org_k,
                     float* __restrict__ rot_q, float* __restrict__ rot_k)
{
  __shared__ __bf16 Ah[128 * ROWA];
  __shared__ __bf16 Al[128 * ROWA];
  __shared__ __bf16 Bh[64 * ROWB];
  __shared__ __bf16 Bl[64 * ROWB];

  const int t    = threadIdx.x;
  const int wave = t >> 5;
  const int lane = t & 31;
  const int g    = lane >> 4;      // half-wave selector in WMMA layouts
  const int ln   = lane & 15;

  const int n0 = blockIdx.x * 64;
  const int m0 = blockIdx.y * 128;

  v8f acc[4];
#pragma unroll
  for (int i = 0; i < 4; ++i) acc[i] = (v8f)0.0f;

  for (int k0 = 0; k0 < HID; k0 += 32) {
    __syncthreads();
    // ---- stage A tile (128 x 32) as hi/lo bf16, [m][k] k-contiguous ----
#pragma unroll
    for (int i = 0; i < 4; ++i) {
      int idx = t + 256 * i;                 // 0..1023 float4s
      int row = idx >> 3;
      int c4  = (idx & 7) << 2;
      f4 v = *(const f4*)(Xin + (size_t)(m0 + row) * HID + k0 + c4);
      bf4 hi, lo;
#pragma unroll
      for (int j = 0; j < 4; ++j) {
        __bf16 h = (__bf16)v[j];
        hi[j] = h;
        lo[j] = (__bf16)(v[j] - (float)h);
      }
      *(bf4*)(Ah + row * ROWA + c4) = hi;    // 8B ds_store
      *(bf4*)(Al + row * ROWA + c4) = lo;
    }
    // ---- stage B tile (32 x 64) transposed to [n][k], hi/lo bf16 ----
#pragma unroll
    for (int i = 0; i < 2; ++i) {
      int idx = t + 256 * i;                 // 0..511 float4s
      int kr  = idx >> 4;
      int c4  = (idx & 15) << 2;
      f4 v = *(const f4*)(Wm + (size_t)(k0 + kr) * NQK + n0 + c4);
#pragma unroll
      for (int j = 0; j < 4; ++j) {
        __bf16 h = (__bf16)v[j];
        Bh[(c4 + j) * ROWB + kr] = h;
        Bl[(c4 + j) * ROWB + kr] = (__bf16)(v[j] - (float)h);
      }
    }
    __syncthreads();

    // ---- A fragment for this wave's 16 rows (16x32 bf16 layout) ----
    const __bf16* arh = Ah + (wave * 16 + ln) * ROWA;
    const __bf16* arl = Al + (wave * 16 + ln) * ROWA;
    v16bf afh = cat8(*(const bf8*)(arh + 16 * g), *(const bf8*)(arh + 16 * g + 8));
    v16bf afl = cat8(*(const bf8*)(arl + 16 * g), *(const bf8*)(arl + 16 * g + 8));

#pragma unroll
    for (int tn = 0; tn < 4; ++tn) {
      const __bf16* brh = Bh + (tn * 16 + ln) * ROWB + 16 * g;
      const __bf16* brl = Bl + (tn * 16 + ln) * ROWB + 16 * g;
      v16bf bfh = cat8(*(const bf8*)brh, *(const bf8*)(brh + 8));
      v16bf bfl = cat8(*(const bf8*)brl, *(const bf8*)(brl + 8));
      acc[tn] = wmma_bf16(afh, bfh, acc[tn]);   // hi*hi
      acc[tn] = wmma_bf16(afh, bfl, acc[tn]);   // hi*lo
      acc[tn] = wmma_bf16(afl, bfh, acc[tn]);   // lo*hi
    }
  }

  // ---- epilogue: bias, split q/k, RoPE, scatter outputs ----
#pragma unroll
  for (int tn = 0; tn < 4; ++tn) {
    const int ng  = n0 + tn * 16 + ln;     // global column, fixed per lane
    const float bv = bias[ng];
    const int h   = ng >> 7;               // head
    const int c   = ng & 127;
    const int isk = c >> 6;                // 0 = q-part, 1 = k-part
    const int d   = c & 63;                // rotary dim
    const float inv = __expf(-(float)(d & 31) * ROPE_LN);
    float* orgp = isk ? org_k : org_q;
    float* rotp = isk ? rot_k : rot_q;
#pragma unroll
    for (int r = 0; r < 8; ++r) {
      float val = acc[tn][r] + bv;
      // adjacent head-dim (d^1) lives in the adjacent lane (n differs by 1)
      float partner = __shfl_xor(val, 1, 32);
      const int mg = m0 + wave * 16 + r + 8 * g;  // global row (b*S + s)
      const int s  = mg & (SEQ - 1);
      float cs, sn;
      __sincosf((float)s * inv, &sn, &cs);
      // careful: __sincosf(x, &s, &c) -> first is sin
      float w2  = (d & 1) ? partner : -partner;   // w2[2i]=-w[2i+1], w2[2i+1]=w[2i]
      float rot = val * cs + w2 * sn;
      const size_t oi = ((size_t)mg * HEADS + h) * HEAD + d;
      orgp[oi] = val;
      rotp[oi] = rot;
    }
  }
}

// ---------------------------------------------------------------------------
// Kernel 2: logits[b,m,n,h] = (q̂[b,m,h,:]·k̂[b,n,h,:]) / 8
// Block = 256 thr (8 waves) owns (b, 16-row m-tile, 128-col n-tile) for ALL
// 12 heads.  q̂ tile staged once in LDS (reused by 8 waves x 12 heads);
// k̂ fragments streamed from global.  Split-bf16 WMMA, K=64 in two 32-steps.
// ---------------------------------------------------------------------------
constexpr int QROW = HEADS * HEAD + 4;   // 772 floats: padded, 16B-aligned rows

__global__ __launch_bounds__(256)
void logits_kernel(const float* __restrict__ rot_q,
                   const float* __restrict__ rot_k,
                   float* __restrict__ out)
{
  __shared__ float qs[16 * QROW];        // 49,408 B

  const int t    = threadIdx.x;
  const int wave = t >> 5;
  const int lane = t & 31;
  const int g    = lane >> 4;
  const int ln   = lane & 15;

  const int n0 = blockIdx.x * 128;
  const int m0 = blockIdx.y * 16;
  const int b  = blockIdx.z;

  // ---- stage q̂ tile: 16 rows x (12 heads * 64) fp32, contiguous per row ----
  const float* qbase = rot_q + ((size_t)(b * SEQ + m0)) * (HEADS * HEAD);
#pragma unroll
  for (int i = 0; i < 12; ++i) {
    int idx = t + 256 * i;               // 0..3071 float4s (16 rows * 192)
    int m = idx / 192;
    int c = idx % 192;
    f4 v = *(const f4*)(qbase + (size_t)m * (HEADS * HEAD) + c * 4);
    *(f4*)(qs + m * QROW + c * 4) = v;
  }
  __syncthreads();

  const int nw = n0 + wave * 16 + ln;    // this lane's key row (column of logits)
  const float* kbase = rot_k + ((size_t)(b * SEQ + nw)) * (HEADS * HEAD);

  for (int h = 0; h < HEADS; ++h) {
    v8f acc = (v8f)0.0f;
#pragma unroll
    for (int kk = 0; kk < 2; ++kk) {
      // A fragment (16x32 bf16): lane holds row m=ln, K contiguous 16 at 16g
      const float* ap = qs + ln * QROW + h * HEAD + kk * 32;
      f4 aa[4];
      aa[0] = *(const f4*)(ap + 16 * g);
      aa[1] = *(const f4*)(ap + 16 * g + 4);
      aa[2] = *(const f4*)(ap + 16 * g + 8);
      aa[3] = *(const f4*)(ap + 16 * g + 12);
      v16bf afh, afl;
#pragma unroll
      for (int j = 0; j < 16; ++j) {
        float x = aa[j >> 2][j & 3];
        __bf16 hi = (__bf16)x;
        afh[j] = hi;
        afl[j] = (__bf16)(x - (float)hi);
      }
      // B fragment (32x16 bf16): lane holds col n=nw, K contiguous 16 at 16g
      const float* bp = kbase + h * HEAD + kk * 32 + 16 * g;
      f4 bb[4];
      bb[0] = *(const f4*)(bp + 0);
      bb[1] = *(const f4*)(bp + 4);
      bb[2] = *(const f4*)(bp + 8);
      bb[3] = *(const f4*)(bp + 12);
      v16bf bfh, bfl;
#pragma unroll
      for (int j = 0; j < 16; ++j) {
        float x = bb[j >> 2][j & 3];
        __bf16 hi = (__bf16)x;
        bfh[j] = hi;
        bfl[j] = (__bf16)(x - (float)hi);
      }
      acc = wmma_bf16(afh, bfh, acc);
      acc = wmma_bf16(afh, bfl, acc);
      acc = wmma_bf16(afl, bfh, acc);
    }
    // ---- store scaled logits in (b,m,n,h) layout; block collectively covers
    //      contiguous 1536-float rows -> coalesces in L2 before HBM ----
#pragma unroll
    for (int r = 0; r < 8; ++r) {
      const int mg = m0 + r + 8 * g;
      out[(((size_t)(b * SEQ + mg)) * SEQ + nw) * HEADS + h] = acc[r] * 0.125f;
    }
  }
}

extern "C" void kernel_launch(void* const* d_in, const int* in_sizes, int n_in,
                              void* d_out, int out_size, void* d_ws, size_t ws_size,
                              hipStream_t stream) {
  const float* Xin  = (const float*)d_in[0];   // (4,1024,768)
  const float* Wm   = (const float*)d_in[1];   // (768,1536)
  const float* bias = (const float*)d_in[2];   // (1536,)
  float* out   = (float*)d_out;                // logits | org_qw | org_kw
  float* org_q = out + LOGITS_ELEMS;
  float* org_k = org_q + QK_ELEMS;
  float* rot_q = (float*)d_ws;                 // rotated q (fp32)
  float* rot_k = rot_q + QK_ELEMS;             // rotated k (fp32)

  dim3 g1(NQK / 64, (BSZ * SEQ) / 128);        // 24 x 32 blocks
  qkv_rope_kernel<<<g1, 256, 0, stream>>>(Xin, Wm, bias, org_q, org_k, rot_q, rot_k);

  dim3 g2(SEQ / 128, SEQ / 16, BSZ);           // 8 x 64 x 4 blocks
  logits_kernel<<<g2, 256, 0, stream>>>(rot_q, rot_k, out);
}
